// GPT_70858370449923
// MI455X (gfx1250) — matrix operand order repaired and checked by reading, not compiled, not run on hardware
//
#include <hip/hip_runtime.h>

// ---------------------------------------------------------------------------
// MI455X (gfx1250) GPT forward. All GEMMs via v_wmma_f32_16x16x32_bf16.
// Weights are transposed+converted to bf16 [N][K] once per launch so every
// weight GEMM stages both A and B tiles as contiguous rows via
// global_load_async_to_lds_b128 (ASYNCcnt) into double-buffered, padded LDS.
// Only the P x V GEMM uses the transpose-on-store path.
// ---------------------------------------------------------------------------

typedef __bf16 bf16_t;
typedef bf16_t v16bf __attribute__((ext_vector_type(16)));
typedef bf16_t v8bf  __attribute__((ext_vector_type(8)));
typedef bf16_t v2bf  __attribute__((ext_vector_type(2)));
typedef float  v8f   __attribute__((ext_vector_type(8)));
typedef int    v4i_vs __attribute__((vector_size(16)));   // builtin's expected pointee

namespace cfg {
constexpr int KD  = 768;
constexpr int H   = 8;
constexpr int L   = 2;
constexpr int VOC = 32000;
constexpr int B   = 2;
constexpr int T   = 1024;
constexpr int M   = B * T;      // 2048 rows of activations
constexpr int KH  = KD * H;     // 6144
constexpr int FFD = 4 * KD;     // 3072
}

#define BM 128
#define BN 128
#define BK 32
#define NT 256
#define PAD 40   // LDS row stride in elements (80B) -> conflict-free fragment loads

// ---------------------------------------------------------------------------
// CDNA5 async memory->LDS copy (ASYNCcnt) with safe fallback
// ---------------------------------------------------------------------------
__device__ __forceinline__ void async_copy16(const bf16_t* g, bf16_t* l) {
#if defined(__has_builtin) && __has_builtin(__builtin_amdgcn_global_load_async_to_lds_b128)
  __builtin_amdgcn_global_load_async_to_lds_b128(
      (__attribute__((address_space(1))) v4i_vs*)g,
      (__attribute__((address_space(3))) v4i_vs*)l, 0, 0);
#else
  *(v8bf*)l = *(const v8bf*)g;
#endif
}
__device__ __forceinline__ void wait_async0() {
#if defined(__has_builtin) && __has_builtin(__builtin_amdgcn_s_wait_asynccnt)
  __builtin_amdgcn_s_wait_asynccnt(0);
#else
  asm volatile("s_wait_asynccnt 0" ::: "memory");
#endif
}

// ---------------------------------------------------------------------------
// Reduction helpers (wave32)
// ---------------------------------------------------------------------------
__device__ __forceinline__ float wave_sum(float v) {
#pragma unroll
  for (int off = 16; off > 0; off >>= 1) v += __shfl_down(v, off, 32);
  return v;
}
__device__ __forceinline__ float wave_max(float v) {
#pragma unroll
  for (int off = 16; off > 0; off >>= 1) v = fmaxf(v, __shfl_down(v, off, 32));
  return v;
}

// ---------------------------------------------------------------------------
// Tiled transpose + fp32->bf16: in[R][C] -> out[C][R].  Coalesced both sides.
// grid = (C/32, R/32, layers), block = 256 (32x8).
// ---------------------------------------------------------------------------
__global__ __launch_bounds__(256) void transpose_cvt(const float* __restrict__ in,
                                                     bf16_t* __restrict__ out,
                                                     int R, int C) {
  __shared__ float tile[32][33];
  const long zoff = (long)blockIdx.z * R * C;
  const int c0 = blockIdx.x * 32, r0 = blockIdx.y * 32;
  const int tx = threadIdx.x & 31, ty = threadIdx.x >> 5;
#pragma unroll
  for (int i = 0; i < 32; i += 8)
    tile[ty + i][tx] = in[zoff + (long)(r0 + ty + i) * C + c0 + tx];
  __syncthreads();
#pragma unroll
  for (int i = 0; i < 32; i += 8)
    out[zoff + (long)(c0 + ty + i) * R + r0 + tx] = (bf16_t)tile[tx][ty + i];
}

// ---------------------------------------------------------------------------
// Embedding + sinusoidal positional encoding -> bf16 activations
// ---------------------------------------------------------------------------
__global__ __launch_bounds__(256) void embed_pos_kernel(const int* __restrict__ x,
                                                        const float* __restrict__ emb,
                                                        bf16_t* __restrict__ out) {
  using namespace cfg;
  const int row = blockIdx.x;          // 0..M-1
  const int t   = row % T;
  const int tok = x[row];
  for (int k = threadIdx.x; k < KD; k += 256) {
    float ex  = 2.0f * (float)(k & ~1) / (float)KD;
    float div = __powf(10000.0f, ex);
    float ang = (float)t / div;
    float pos = (k & 1) ? __cosf(ang) : __sinf(ang);
    out[(long)row * KD + k] = (bf16_t)(emb[(long)tok * KD + k] + pos);
  }
}

// ---------------------------------------------------------------------------
// WMMA GEMM:  C[M,N] = A[M,Kd] * op(B) (+ bias) (+ GELU)
//   BT=true : B is [N,Kd] row-major (used as B^T) -> fully async-staged.
//   BT=false: B is [Kd,N] row-major -> register-staged, transposed ds stores.
// Batched over blockIdx.z with (b,h) = (z/Hz, z%Hz) pointer strides (elements).
// Outputs: Cf (fp32) and/or Cb (bf16), either may be null.
// All dims are multiples of the tile sizes for every GEMM in this model.
// ---------------------------------------------------------------------------
template <bool BT, bool ACT_GELU>
__global__ __launch_bounds__(NT) void gemm_bf16_wmma(
    const bf16_t* __restrict__ A, int lda, long sAb, long sAh,
    const bf16_t* __restrict__ Bm, int ldb, long sBb, long sBh,
    const float* __restrict__ bias,
    float* __restrict__ Cf, bf16_t* __restrict__ Cb, int ldc, long sCb_, long sCh,
    int Mdim, int Ndim, int Kdim, int Hz) {
  __shared__ bf16_t sA[2][BM * PAD];    // [m][k], padded
  __shared__ bf16_t sBt[2][BN * PAD];   // [n][k], padded

  const int z  = blockIdx.z;
  const int zb = z / Hz, zh = z % Hz;
  A  += zb * sAb + zh * sAh;
  Bm += zb * sBb + zh * sBh;
  const long coff = zb * sCb_ + zh * sCh;

  const int tid   = threadIdx.x;
  const int lane  = tid & 31;
  const int wave  = tid >> 5;
  const int wr    = wave >> 1;          // 0..3 : 32-row slab
  const int wc    = wave & 1;           // 0..1 : 64-col slab
  const int m0    = blockIdx.y * BM;
  const int n0    = blockIdx.x * BN;
  const int lhalf = (lane < 16) ? 0 : 1;
  const int lm    = lane & 15;

  // staging indices: 128 rows x 32 cols per tile, 2x b128 per thread
  const int arow = tid >> 1;            // 0..127
  const int acol = (tid & 1) * 16;      // 0 / 16
  const int kp   = tid >> 4;            // 0..15 : k-row pair 2kp,2kp+1 (!BT)
  const int nnp  = (tid & 15) * 8;      // 0..120 (!BT)

  // ---- async staging: A always; B too in BT mode ----
  auto stage_async = [&](int buf, int k0) {
    const bf16_t* ap = A + (long)(m0 + arow) * lda + k0 + acol;
    bf16_t* la = &sA[buf][arow * PAD + acol];
    async_copy16(ap, la);
    async_copy16(ap + 8, la + 8);
    if (BT) {
      const bf16_t* bp = Bm + (long)(n0 + arow) * ldb + k0 + acol;
      bf16_t* lb = &sBt[buf][arow * PAD + acol];
      async_copy16(bp, lb);
      async_copy16(bp + 8, lb + 8);
    }
  };
  // ---- !BT B staging: registers + packed-pair transposed stores ----
  v8bf gb0, gb1;
  auto gloadB = [&](int k0) {
    const bf16_t* bp = Bm + (long)(k0 + 2 * kp) * ldb + n0 + nnp;
    gb0 = *(const v8bf*)(bp);
    gb1 = *(const v8bf*)(bp + ldb);
  };
  auto sstoreB = [&](int buf) {
#pragma unroll
    for (int i = 0; i < 8; ++i) {       // packed bf16 pairs -> ds_store_b32
      v2bf pr = {gb0[i], gb1[i]};
      *(v2bf*)(&sBt[buf][(nnp + i) * PAD + 2 * kp]) = pr;
    }
  };

  v8f acc[2][4];
#pragma unroll
  for (int i = 0; i < 2; ++i)
#pragma unroll
    for (int j = 0; j < 4; ++j)
#pragma unroll
      for (int r = 0; r < 8; ++r) acc[i][j][r] = 0.0f;

  stage_async(0, 0);
  if (!BT) { gloadB(0); sstoreB(0); }
  wait_async0();
  int cur = 0;
  for (int k0 = 0; k0 < Kdim; k0 += BK) {
    __syncthreads();                       // LDS[cur] published to all waves
    const bool more = (k0 + BK) < Kdim;
    if (more) {
      stage_async(cur ^ 1, k0 + BK);       // DMA next tile during WMMAs
      if (!BT) gloadB(k0 + BK);
    }

    // ---- fragments (ISA 16-bit A 16x32 / B 32x16 lane layouts) ----
    v16bf afrag[2], bfrag[4];
#pragma unroll
    for (int i = 0; i < 2; ++i) {
      const bf16_t* pA = &sA[cur][(wr * 32 + i * 16 + lm) * PAD + lhalf * 8];
      v8bf lo = *(const v8bf*)(pA);
      v8bf hi = *(const v8bf*)(pA + 16);
#pragma unroll
      for (int e = 0; e < 8; ++e) { afrag[i][e] = lo[e]; afrag[i][8 + e] = hi[e]; }
    }
#pragma unroll
    for (int j = 0; j < 4; ++j) {
      const bf16_t* pB = &sBt[cur][(wc * 64 + j * 16 + lm) * PAD + lhalf * 8];
      v8bf lo = *(const v8bf*)(pB);
      v8bf hi = *(const v8bf*)(pB + 16);
#pragma unroll
      for (int e = 0; e < 8; ++e) { bfrag[j][e] = lo[e]; bfrag[j][8 + e] = hi[e]; }
    }

#pragma unroll
    for (int i = 0; i < 2; ++i)
#pragma unroll
      for (int j = 0; j < 4; ++j)
        acc[i][j] = __builtin_amdgcn_wmma_f32_16x16x32_bf16(
            false, afrag[i], false, bfrag[j], (short)0, acc[i][j], false, false);

    if (more) {
      if (!BT) sstoreB(cur ^ 1);
      wait_async0();                       // next buffer complete before barrier
    }
    cur ^= 1;
  }

  // ---- epilogue: C/D layout: VGPR r -> M = r + (lane<16?0:8), N = lane&15 ----
#pragma unroll
  for (int i = 0; i < 2; ++i) {
#pragma unroll
    for (int j = 0; j < 4; ++j) {
      int n = n0 + wc * 64 + j * 16 + lm;
      float bv = bias ? bias[n] : 0.0f;
#pragma unroll
      for (int r = 0; r < 8; ++r) {
        int m = m0 + wr * 32 + i * 16 + r + lhalf * 8;
        float v = acc[i][j][r] + bv;
        if (ACT_GELU) v = 0.5f * v * (1.0f + erff(v * 0.70710678118654752f));
        long idx = coff + (long)m * ldc + n;
        if (Cf) Cf[idx] = v;
        if (Cb) Cb[idx] = (bf16_t)v;
      }
    }
  }
}

// ---------------------------------------------------------------------------
// Row softmax over scores S[z][row][T] (scale folded in), writes bf16 P.
// ---------------------------------------------------------------------------
__global__ __launch_bounds__(256) void softmax_rows(const float* __restrict__ S,
                                                    bf16_t* __restrict__ P) {
  using namespace cfg;
  const float scale = 0.03608439182435161f;  // 1/sqrt(768)
  const long base = ((long)blockIdx.y * T + blockIdx.x) * T;
  const float* s = S + base;
  bf16_t* p = P + base;
  const int tid = threadIdx.x;

  __shared__ float red[8];
  float vals[4];
  float m = -3.0e38f;
#pragma unroll
  for (int i = 0; i < 4; ++i) {
    vals[i] = s[tid + i * 256] * scale;
    m = fmaxf(m, vals[i]);
  }
  m = wave_max(m);
  if ((tid & 31) == 0) red[tid >> 5] = m;
  __syncthreads();
  float mall = red[0];
#pragma unroll
  for (int w = 1; w < 8; ++w) mall = fmaxf(mall, red[w]);
  __syncthreads();

  float sum = 0.0f;
#pragma unroll
  for (int i = 0; i < 4; ++i) {
    vals[i] = __expf(vals[i] - mall);
    sum += vals[i];
  }
  sum = wave_sum(sum);
  if ((tid & 31) == 0) red[tid >> 5] = sum;
  __syncthreads();
  float stot = 0.0f;
#pragma unroll
  for (int w = 0; w < 8; ++w) stot += red[w];
  float inv = 1.0f / stot;
#pragma unroll
  for (int i = 0; i < 4; ++i) p[tid + i * 256] = (bf16_t)(vals[i] * inv);
}

// ---------------------------------------------------------------------------
// Row LayerNorm over KD=768 columns. grid = rows, block = 256, 3 elems/thread.
// ---------------------------------------------------------------------------
__global__ __launch_bounds__(256) void layernorm_rows(const float* __restrict__ X,
                                                      const float* __restrict__ g,
                                                      const float* __restrict__ b,
                                                      bf16_t* __restrict__ Ob,
                                                      float* __restrict__ Of) {
  using namespace cfg;
  const long base = (long)blockIdx.x * KD;
  const float* x = X + base;
  const int tid = threadIdx.x;

  __shared__ float red[8];
  float v[3];
  float s = 0.0f;
#pragma unroll
  for (int i = 0; i < 3; ++i) { v[i] = x[tid + i * 256]; s += v[i]; }
  s = wave_sum(s);
  if ((tid & 31) == 0) red[tid >> 5] = s;
  __syncthreads();
  float tot = 0.0f;
#pragma unroll
  for (int w = 0; w < 8; ++w) tot += red[w];
  const float mu = tot * (1.0f / KD);
  __syncthreads();

  float d2 = 0.0f;
#pragma unroll
  for (int i = 0; i < 3; ++i) { float d = v[i] - mu; d2 += d * d; }
  d2 = wave_sum(d2);
  if ((tid & 31) == 0) red[tid >> 5] = d2;
  __syncthreads();
  float vtot = 0.0f;
#pragma unroll
  for (int w = 0; w < 8; ++w) vtot += red[w];
  const float inv = rsqrtf(vtot * (1.0f / KD) + 1e-5f);

#pragma unroll
  for (int i = 0; i < 3; ++i) {
    int k = tid + i * 256;
    float y = (v[i] - mu) * inv * g[k] + b[k];
    if (Ob) Ob[base + k] = (bf16_t)y;
    if (Of) Of[base + k] = y;
  }
}

// ---------------------------------------------------------------------------
// Host orchestration
// ---------------------------------------------------------------------------
static inline void* wsal(char*& p, size_t bytes) {
  void* r = (void*)p;
  p += (bytes + 255) & ~(size_t)255;
  return r;
}

extern "C" void kernel_launch(void* const* d_in, const int* in_sizes, int n_in,
                              void* d_out, int out_size, void* d_ws, size_t ws_size,
                              hipStream_t stream) {
  using namespace cfg;
  const int*   x    = (const int*)d_in[0];
  const float* emb  = (const float*)d_in[1];
  const float* Wq   = (const float*)d_in[2];
  const float* Wk   = (const float*)d_in[3];
  const float* Wv   = (const float*)d_in[4];
  const float* Wu   = (const float*)d_in[5];
  const float* bu   = (const float*)d_in[6];
  const float* ln1g = (const float*)d_in[7];
  const float* ln1b = (const float*)d_in[8];
  const float* Wf1  = (const float*)d_in[9];
  const float* bf1  = (const float*)d_in[10];
  const float* Wf2  = (const float*)d_in[11];
  const float* bf2  = (const float*)d_in[12];
  const float* ln2g = (const float*)d_in[13];
  const float* ln2b = (const float*)d_in[14];
  const float* lnfg = (const float*)d_in[15];
  const float* lnfb = (const float*)d_in[16];
  const float* Wout = (const float*)d_in[17];
  const float* bout = (const float*)d_in[18];
  float* out = (float*)d_out;

  // ---- workspace layout (all weights transposed: [N][K] bf16) ----
  char* p = (char*)d_ws;
  bf16_t* wqtb   = (bf16_t*)wsal(p, (size_t)L * KD * KH * 2);   // [KH][KD] per layer
  bf16_t* wktb   = (bf16_t*)wsal(p, (size_t)L * KD * KH * 2);
  bf16_t* wvtb   = (bf16_t*)wsal(p, (size_t)L * KD * KH * 2);
  bf16_t* wutb   = (bf16_t*)wsal(p, (size_t)L * KH * KD * 2);   // [KD][KH] per layer
  bf16_t* wf1tb  = (bf16_t*)wsal(p, (size_t)L * KD * FFD * 2);  // [FFD][KD]
  bf16_t* wf2tb  = (bf16_t*)wsal(p, (size_t)L * FFD * KD * 2);  // [KD][FFD]
  bf16_t* wouttb = (bf16_t*)wsal(p, (size_t)KD * VOC * 2);      // [VOC][KD]
  bf16_t* xebf   = (bf16_t*)wsal(p, (size_t)M * KD * 2);
  bf16_t* qb     = (bf16_t*)wsal(p, (size_t)M * KH * 2);
  bf16_t* kb     = (bf16_t*)wsal(p, (size_t)M * KH * 2);
  bf16_t* vb     = (bf16_t*)wsal(p, (size_t)M * KH * 2);
  float*  Sf     = (float*) wsal(p, (size_t)B * H * T * T * 4);
  bf16_t* Pb     = (bf16_t*)wsal(p, (size_t)B * H * T * T * 2);
  float*  attf   = (float*) wsal(p, (size_t)M * KD * 4);
  bf16_t* anb    = (bf16_t*)wsal(p, (size_t)M * KD * 2);
  float*  fff    = (float*) wsal(p, (size_t)M * KD * 4);
  float*  xef    = (float*) wsal(p, (size_t)M * KD * 4);
  bf16_t* xfb    = (bf16_t*)wsal(p, (size_t)M * KD * 2);
  bf16_t* ybf    = qb;   // q dead after scores GEMM -> reuse
  bf16_t* hb     = kb;   // k dead after scores GEMM -> reuse

  // ---- weight transpose + bf16 conversion ----
  const dim3 tb(256);
  transpose_cvt<<<dim3(KH / 32, KD / 32, L), tb, 0, stream>>>(Wq,  wqtb,  KD,  KH);
  transpose_cvt<<<dim3(KH / 32, KD / 32, L), tb, 0, stream>>>(Wk,  wktb,  KD,  KH);
  transpose_cvt<<<dim3(KH / 32, KD / 32, L), tb, 0, stream>>>(Wv,  wvtb,  KD,  KH);
  transpose_cvt<<<dim3(KD / 32, KH / 32, L), tb, 0, stream>>>(Wu,  wutb,  KH,  KD);
  transpose_cvt<<<dim3(FFD / 32, KD / 32, L), tb, 0, stream>>>(Wf1, wf1tb, KD,  FFD);
  transpose_cvt<<<dim3(KD / 32, FFD / 32, L), tb, 0, stream>>>(Wf2, wf2tb, FFD, KD);
  transpose_cvt<<<dim3(VOC / 32, KD / 32, 1), tb, 0, stream>>>(Wout, wouttb, KD, VOC);

  // ---- embedding + positional encoding ----
  embed_pos_kernel<<<dim3(M), dim3(256), 0, stream>>>(x, emb, xebf);

  for (int l = 0; l < L; ++l) {
    const bf16_t* wq_l  = wqtb  + (long)l * KD * KH;
    const bf16_t* wk_l  = wktb  + (long)l * KD * KH;
    const bf16_t* wv_l  = wvtb  + (long)l * KD * KH;
    const bf16_t* wu_l  = wutb  + (long)l * KH * KD;
    const bf16_t* wf1_l = wf1tb + (long)l * KD * FFD;
    const bf16_t* wf2_l = wf2tb + (long)l * FFD * KD;

    // Q/K/V projections: [2048x768] x [768x6144] (Wt: [6144][768]) -> bf16
    gemm_bf16_wmma<true, false><<<dim3(KH / BN, M / BM, 1), dim3(NT), 0, stream>>>(
        xebf, KD, 0, 0, wq_l, KD, 0, 0, nullptr, nullptr, qb, KH, 0, 0, M, KH, KD, 1);
    gemm_bf16_wmma<true, false><<<dim3(KH / BN, M / BM, 1), dim3(NT), 0, stream>>>(
        xebf, KD, 0, 0, wk_l, KD, 0, 0, nullptr, nullptr, kb, KH, 0, 0, M, KH, KD, 1);
    gemm_bf16_wmma<true, false><<<dim3(KH / BN, M / BM, 1), dim3(NT), 0, stream>>>(
        xebf, KD, 0, 0, wv_l, KD, 0, 0, nullptr, nullptr, vb, KH, 0, 0, M, KH, KD, 1);

    // Scores: S[b,h] = Q_bh x K_bh^T  (batched z = b*H+h, K is [T][768] = B^T)
    gemm_bf16_wmma<true, false><<<dim3(T / BN, T / BM, B * H), dim3(NT), 0, stream>>>(
        qb, KH, (long)T * KH, KD,
        kb, KH, (long)T * KH, KD,
        nullptr, Sf, nullptr, T, (long)H * T * T, (long)T * T,
        T, T, KD, H);

    // Softmax rows (scale folded), fp32 -> bf16 P
    softmax_rows<<<dim3(T, B * H), dim3(256), 0, stream>>>(Sf, Pb);

    // Y[b,h] = P_bh x V_bh  -> packed heads bf16 [2048 x 6144]  (!BT path)
    gemm_bf16_wmma<false, false><<<dim3(KD / BN, T / BM, B * H), dim3(NT), 0, stream>>>(
        Pb, T, (long)H * T * T, (long)T * T,
        vb, KH, (long)T * KH, KD,
        nullptr, nullptr, ybf, KH, (long)T * KH, KD,
        T, KD, T, H);

    // att = Y x Wu + bu -> fp32  (Wut: [768][6144])
    gemm_bf16_wmma<true, false><<<dim3(KD / BN, M / BM, 1), dim3(NT), 0, stream>>>(
        ybf, KH, 0, 0, wu_l, KH, 0, 0, bu + (long)l * KD,
        attf, nullptr, KD, 0, 0, M, KD, KH, 1);

    // LN1 -> bf16
    layernorm_rows<<<dim3(M), dim3(256), 0, stream>>>(
        attf, ln1g + (long)l * KD, ln1b + (long)l * KD, anb, nullptr);

    // FF1 + exact GELU -> bf16 hidden  (Wf1t: [3072][768])
    gemm_bf16_wmma<true, true><<<dim3(FFD / BN, M / BM, 1), dim3(NT), 0, stream>>>(
        anb, KD, 0, 0, wf1_l, KD, 0, 0, bf1 + (long)l * FFD,
        nullptr, hb, FFD, 0, 0, M, FFD, KD, 1);

    // FF2 -> fp32  (Wf2t: [768][3072])
    gemm_bf16_wmma<true, false><<<dim3(KD / BN, M / BM, 1), dim3(NT), 0, stream>>>(
        hb, FFD, 0, 0, wf2_l, FFD, 0, 0, bf2 + (long)l * KD,
        fff, nullptr, KD, 0, 0, M, KD, FFD, 1);

    // LN2 -> next xe (bf16 for GEMMs, fp32 for final LN)
    layernorm_rows<<<dim3(M), dim3(256), 0, stream>>>(
        fff, ln2g + (long)l * KD, ln2b + (long)l * KD, xebf, xef);
  }

  // Final LN -> bf16
  layernorm_rows<<<dim3(M), dim3(256), 0, stream>>>(xef, lnfg, lnfb, xfb, nullptr);

  // Logits: [2048x768] x [768x32000] (Woutt: [32000][768]) + bout -> fp32 d_out
  gemm_bf16_wmma<true, false><<<dim3(VOC / BN, M / BM, 1), dim3(NT), 0, stream>>>(
      xfb, KD, 0, 0, wouttb, KD, 0, 0, bout,
      out, nullptr, VOC, 0, 0, M, VOC, KD, 1);

  (void)in_sizes; (void)n_in; (void)out_size; (void)ws_size;
}
